// MFCI_model_20959440404536
// MI455X (gfx1250) — matrix operand order repaired and checked by reading, not compiled, not run on hardware
//
#include <hip/hip_runtime.h>
#include <math.h>

// ---------------------------------------------------------------------------
// Types for CDNA5 WMMA (gfx1250, wave32)
// ---------------------------------------------------------------------------
typedef __attribute__((ext_vector_type(16))) __bf16 bf16x16;
typedef __attribute__((ext_vector_type(8)))  float  f32x8;

__device__ __forceinline__ unsigned short bfbits(float f) {
  unsigned u = __builtin_bit_cast(unsigned, f);
  return (unsigned short)((u + 0x7FFFu + ((u >> 16) & 1u)) >> 16);   // RNE
}
__device__ __forceinline__ unsigned pack2bf(float lo, float hi) {
  return (unsigned)bfbits(lo) | ((unsigned)bfbits(hi) << 16);
}
__device__ __forceinline__ __bf16 tobf(float f) {
  return __builtin_bit_cast(__bf16, bfbits(f));
}

// ---------------------------------------------------------------------------
// General batched, strided GEMM:  C = act(alpha * A@B + bias + R)
//   A(m,k) at A[b2*aB2 + b1*aB1 + m*aR + k*aC]     (f32 -> bf16)
//   B(k,n) at Bm[b2*bB2 + b1*bB1 + k*bR + n*bC]    (f32 -> bf16)
//   C(m,n) at C[b2*cB2 + b1*cB1 + m*cR + n]
//   z = blockIdx.z ; b2 = z / bdiv ; b1 = z % bdiv
// Block: 128 threads = 4 waves. Block tile 64(M) x 32(N), K step 64.
// LDS tiles are kept in *fragment-major* bf16 layout so each lane's WMMA
// fragment is 32 contiguous bytes (2x ds_load_b128).
//   A frag slot (ISA 7.12.2): lane = (m&15) + 16*((k>>3)&1), e = 8*(k>>4)+(k&7)
//   B frag slot:              lane = (n&15) + 16*((k>>3)&1), e = 8*(k>>4)+(k&7)
// ---------------------------------------------------------------------------
struct GemmP {
  const float* A; const float* Bm; float* C;
  const float* bias; const float* R;
  long long aB2, aB1, aR, aC;
  long long bB2, bB1, bR, bC;
  long long cB2, cB1, cR;
  long long rB2, rB1, rR;
  int M, Nn, K, bdiv;
  float alpha; int act;   // 0=none 1=relu 2=gelu(erf)
};

__launch_bounds__(128)
__global__ void k_gemm(GemmP p) {
  __shared__ __align__(32) unsigned AsU[2][4][32][8]; // [ktile][mtile][lane][pair] bf16x2
  __shared__ __align__(32) unsigned BsU[2][2][32][8]; // [ktile][ntile][lane][pair] bf16x2
  const int tid  = threadIdx.x;
  const int wave = tid >> 5;
  const int lane = tid & 31;
  const int z  = blockIdx.z;
  const int b2 = z / p.bdiv, b1 = z % p.bdiv;
  const long long aBase = (long long)b2 * p.aB2 + (long long)b1 * p.aB1;
  const long long bBase = (long long)b2 * p.bB2 + (long long)b1 * p.bB1;
  const int m0 = blockIdx.x * 64;
  const int n0 = blockIdx.y * 32;

  f32x8 acc0 = {}, acc1 = {};
  for (int k0 = 0; k0 < p.K; k0 += 64) {
    const bool fast = (k0 + 64 <= p.K);          // block-uniform scalar branch
    // ---- A tile: 64 rows x 64 k, written as bf16 pairs into fragment slots
    #pragma unroll
    for (int i = 0; i < 16; ++i) {
      int f = tid + i * 128;                     // pair id 0..2047
      int r = f >> 5;                            // 0..63
      int k = (f & 31) * 2;                      // 0..62 (even)
      int m = m0 + r; if (m >= p.M) m = p.M - 1; // clamp; stores bounds-checked
      const long long ga = aBase + (long long)m * p.aR + (long long)(k0 + k) * p.aC;
      float v0, v1;
      if (fast) { v0 = p.A[ga]; v1 = p.A[ga + p.aC]; }
      else {
        v0 = (k0 + k     < p.K) ? p.A[ga]         : 0.f;
        v1 = (k0 + k + 1 < p.K) ? p.A[ga + p.aC]  : 0.f;
      }
      int kt = k >> 5, kl = k & 31;
      int ln = (r & 15) + (((kl >> 3) & 1) << 4);
      int e  = ((kl >> 4) << 3) + (kl & 7);      // even
      AsU[kt][r >> 4][ln][e >> 1] = pack2bf(v0, v1);
    }
    // ---- B tile: 64 k x 32 n
    #pragma unroll
    for (int i = 0; i < 8; ++i) {
      int f = tid + i * 128;                     // pair id 0..1023
      int n = f & 31;
      int k = (f >> 5) * 2;                      // 0..62 (even)
      float v0 = 0.f, v1 = 0.f;
      if (n0 + n < p.Nn) {
        const long long gb = bBase + (long long)(k0 + k) * p.bR + (long long)(n0 + n) * p.bC;
        if (fast) { v0 = p.Bm[gb]; v1 = p.Bm[gb + p.bR]; }
        else {
          if (k0 + k     < p.K) v0 = p.Bm[gb];
          if (k0 + k + 1 < p.K) v1 = p.Bm[gb + p.bR];
        }
      }
      int kt = k >> 5, kl = k & 31;
      int ln = (n & 15) + (((kl >> 3) & 1) << 4);
      int e  = ((kl >> 4) << 3) + (kl & 7);      // even
      BsU[kt][n >> 4][ln][e >> 1] = pack2bf(v0, v1);
    }
    __syncthreads();

    bf16x16 a0, a1, b00, b01, b10, b11;
    __builtin_memcpy(&a0,  &AsU[0][wave][lane][0], 32);   // 2x ds_load_b128
    __builtin_memcpy(&a1,  &AsU[1][wave][lane][0], 32);
    __builtin_memcpy(&b00, &BsU[0][0][lane][0], 32);
    __builtin_memcpy(&b01, &BsU[0][1][lane][0], 32);
    __builtin_memcpy(&b10, &BsU[1][0][lane][0], 32);
    __builtin_memcpy(&b11, &BsU[1][1][lane][0], 32);

    acc0 = __builtin_amdgcn_wmma_f32_16x16x32_bf16(false, a0, false, b00, (short)0, acc0, false, false);
    acc1 = __builtin_amdgcn_wmma_f32_16x16x32_bf16(false, a0, false, b01, (short)0, acc1, false, false);
    acc0 = __builtin_amdgcn_wmma_f32_16x16x32_bf16(false, a1, false, b10, (short)0, acc0, false, false);
    acc1 = __builtin_amdgcn_wmma_f32_16x16x32_bf16(false, a1, false, b11, (short)0, acc1, false, false);
    __syncthreads();
  }

  // epilogue: C/D layout -> VGPR r holds M = r + 8*(lane>=16), N = lane&15
  const long long cBase = (long long)b2 * p.cB2 + (long long)b1 * p.cB1;
  const long long rBase = (long long)b2 * p.rB2 + (long long)b1 * p.rB1;
  #pragma unroll
  for (int nt = 0; nt < 2; ++nt) {
    const f32x8 acc = nt ? acc1 : acc0;
    const int n = n0 + nt * 16 + (lane & 15);
    #pragma unroll
    for (int r = 0; r < 8; ++r) {
      int m = m0 + wave * 16 + r + ((lane >> 4) << 3);
      if (m < p.M && n < p.Nn) {
        float v = p.alpha * acc[r];
        if (p.bias) v += p.bias[n];
        if (p.R)    v += p.R[rBase + (long long)m * p.rR + n];
        if (p.act == 1)      v = v > 0.f ? v : 0.f;
        else if (p.act == 2) v = 0.5f * v * (1.f + erff(v * 0.70710678118654752f));
        p.C[cBase + (long long)m * p.cR + n] = v;
      }
    }
  }
}

// ---------------------------------------------------------------------------
// Fused flash attention (wave32 WMMA, online softmax).
//   O = softmax(scale * Q K^T) V   per batch slice z=(b2,b1)
//   Q(n,d) at Q[b2*qB2+b1*qB1 + n*qR + d*qC]    (same scheme for K,V)
//   O(n,d) at O[b2*oB2+b1*oB1 + n*oR + d]       (d contiguous, head-merged)
// Block = 128 threads = 4 waves, all on the same (b,h); each wave owns a
// 16-query tile; K/V staged in LDS in B-fragment layout per 32-key tile.
// Per-wave C->A layout transpose of P goes through a private LDS patch.
// hs <= 32 (k-dim zero-padded to 32). Rows are independent: OOB query rows
// are clamped+discarded; OOB key columns masked to -3e38 so exp() -> 0.
// ---------------------------------------------------------------------------
struct FlashP {
  const float* Q; const float* K; const float* V; float* O;
  long long qB2, qB1, qR, qC;
  long long kB2, kB1, kR, kC;
  long long vB2, vB1, vR, vC;
  long long oB2, oB1, oR;
  int Nq, Nk, hs, bdiv;
  float scale;
};

__launch_bounds__(128)
__global__ void k_flash(FlashP p) {
  __shared__ __align__(32) unsigned KtU[2][32][8];   // K tile, B-frag layout (d = k-dim)
  __shared__ __align__(32) unsigned VtU[2][32][8];   // V tile, B-frag layout (key = k-dim)
  __shared__ __align__(32) __bf16  PT[4][32][16];    // per-wave P tile, A-frag layout
  const int tid  = threadIdx.x;
  const int wave = tid >> 5;
  const int lane = tid & 31;
  const int z  = blockIdx.y;
  const int b2 = z / p.bdiv, b1 = z % p.bdiv;
  const long long qBase = (long long)b2 * p.qB2 + (long long)b1 * p.qB1;
  const long long kBase = (long long)b2 * p.kB2 + (long long)b1 * p.kB1;
  const long long vBase = (long long)b2 * p.vB2 + (long long)b1 * p.vB1;
  const int q0 = (blockIdx.x * 4 + wave) * 16;

  // Q fragment (A 16x32), rows clamped, k-dim zero-padded to 32
  bf16x16 aq;
  {
    int m = q0 + (lane & 15); if (m >= p.Nq) m = p.Nq - 1;
    #pragma unroll
    for (int e = 0; e < 16; ++e) {
      int d = ((lane >> 4) << 4) + e;
      float v = (d < p.hs) ? p.Q[qBase + (long long)m * p.qR + (long long)d * p.qC] : 0.f;
      aq[e] = tobf(v);
    }
  }

  float mrun[8], lrun[8];
  #pragma unroll
  for (int r = 0; r < 8; ++r) { mrun[r] = -3.0e38f; lrun[r] = 0.f; }
  f32x8 o0 = {}, o1 = {};

  for (int j0 = 0; j0 < p.Nk; j0 += 32) {
    // ---- cooperative K/V tile staging (512 pairs each / 128 threads) ----
    #pragma unroll
    for (int i = 0; i < 4; ++i) {
      int f = tid + i * 128;                      // 0..511
      // K pair: key j, d even pair (frag: lane=(j&15)+16*(d>>4), e=d&15)
      {
        int j = f & 31;
        int d = (f >> 5) * 2;
        int key = j0 + j; if (key >= p.Nk) key = p.Nk - 1;
        float v0 = 0.f, v1 = 0.f;
        if (d < p.hs) {
          long long g = kBase + (long long)key * p.kR + (long long)d * p.kC;
          v0 = p.K[g];
          if (d + 1 < p.hs) v1 = p.K[g + p.kC];
        }
        KtU[j >> 4][(j & 15) + ((d >> 4) << 4)][(d & 15) >> 1] = pack2bf(v0, v1);
      }
      // V pair: col d, key even pair (frag: lane=(d&15)+16*((key>>3)&1), e=8*(key>>4)+(key&7))
      {
        int d = f & 31;
        int j = (f >> 5) * 2;
        int keyA = j0 + j, keyB = keyA + 1;
        if (keyA >= p.Nk) keyA = p.Nk - 1;
        if (keyB >= p.Nk) keyB = p.Nk - 1;
        float w0 = 0.f, w1 = 0.f;
        if (d < p.hs) {
          w0 = p.V[vBase + (long long)keyA * p.vR + (long long)d * p.vC];
          w1 = p.V[vBase + (long long)keyB * p.vR + (long long)d * p.vC];
        }
        VtU[d >> 4][(d & 15) + (((j >> 3) & 1) << 4)][(((j >> 4) << 3) + (j & 7)) >> 1] = pack2bf(w0, w1);
      }
    }
    __syncthreads();

    bf16x16 bk0, bk1;
    __builtin_memcpy(&bk0, &KtU[0][lane][0], 32);
    __builtin_memcpy(&bk1, &KtU[1][lane][0], 32);
    const f32x8 zro = {};
    f32x8 s0 = __builtin_amdgcn_wmma_f32_16x16x32_bf16(false, aq, false, bk0, (short)0, zro, false, false);
    f32x8 s1 = __builtin_amdgcn_wmma_f32_16x16x32_bf16(false, aq, false, bk1, (short)0, zro, false, false);

    const int col  = lane & 15;
    const bool ok0 = (j0 + col      < p.Nk);
    const bool ok1 = (j0 + col + 16 < p.Nk);
    #pragma unroll
    for (int r = 0; r < 8; ++r) {
      float t0 = ok0 ? s0[r] * p.scale : -3.0e38f;
      float t1 = ok1 ? s1[r] * p.scale : -3.0e38f;
      // half-wave butterfly row reduction (row r in lanes 0-15, r+8 in 16-31)
      float rmax = fmaxf(t0, t1);
      rmax = fmaxf(rmax, __shfl_xor(rmax, 1));
      rmax = fmaxf(rmax, __shfl_xor(rmax, 2));
      rmax = fmaxf(rmax, __shfl_xor(rmax, 4));
      rmax = fmaxf(rmax, __shfl_xor(rmax, 8));
      float mnew = fmaxf(mrun[r], rmax);
      float corr = __expf(mrun[r] - mnew);       // mrun=-3e38 first pass -> 0
      float p0 = __expf(t0 - mnew);              // masked cols -> 0
      float p1 = __expf(t1 - mnew);
      float rsum = p0 + p1;
      rsum += __shfl_xor(rsum, 1);
      rsum += __shfl_xor(rsum, 2);
      rsum += __shfl_xor(rsum, 4);
      rsum += __shfl_xor(rsum, 8);
      mrun[r] = mnew;
      lrun[r] = lrun[r] * corr + rsum;
      o0[r] = o0[r] * corr;
      o1[r] = o1[r] * corr;
      // scatter P (C layout) into per-wave A-frag LDS patch
      int m = r + ((lane >> 4) << 3);            // 0..15
      int kc0 = col, kc1 = col + 16;
      PT[wave][(m & 15) + (((kc0 >> 3) & 1) << 4)][((kc0 >> 4) << 3) + (kc0 & 7)] = tobf(p0);
      PT[wave][(m & 15) + (((kc1 >> 3) & 1) << 4)][((kc1 >> 4) << 3) + (kc1 & 7)] = tobf(p1);
    }

    bf16x16 ap, bv0, bv1;
    __builtin_memcpy(&ap,  &PT[wave][lane][0], 32);      // same-wave LDS, dscnt-ordered
    __builtin_memcpy(&bv0, &VtU[0][lane][0], 32);
    __builtin_memcpy(&bv1, &VtU[1][lane][0], 32);
    o0 = __builtin_amdgcn_wmma_f32_16x16x32_bf16(false, ap, false, bv0, (short)0, o0, false, false);
    if (p.hs > 16)   // grid-uniform branch: EXEC stays all-ones
      o1 = __builtin_amdgcn_wmma_f32_16x16x32_bf16(false, ap, false, bv1, (short)0, o1, false, false);
    __syncthreads();
  }

  // epilogue: O rows m = q0 + r + 8*(lane>=16), cols d = lane&15 (+16)
  const long long oBase = (long long)b2 * p.oB2 + (long long)b1 * p.oB1;
  #pragma unroll
  for (int r = 0; r < 8; ++r) {
    int m = q0 + r + ((lane >> 4) << 3);
    if (m < p.Nq) {
      float inv = 1.f / lrun[r];
      int d0 = lane & 15;
      p.O[oBase + (long long)m * p.oR + d0] = o0[r] * inv;
      if (p.hs > 16)
        p.O[oBase + (long long)m * p.oR + d0 + 16] = o1[r] * inv;
    }
  }
}

// ---------------------------------------------------------------------------
// LayerNorm over rows [rows, D]; optional fused second input (x + x2)
// ---------------------------------------------------------------------------
__launch_bounds__(128)
__global__ void k_layernorm(const float* x, const float* x2, const float* g,
                            const float* b, float* y, int D) {
  __shared__ float red[128];
  const long long row = blockIdx.x;
  const float* xr  = x + row * D;
  const float* x2r = x2 ? x2 + row * D : nullptr;
  float s = 0.f;
  for (int i = threadIdx.x; i < D; i += 128) { float v = xr[i] + (x2r ? x2r[i] : 0.f); s += v; }
  red[threadIdx.x] = s; __syncthreads();
  for (int t = 64; t > 0; t >>= 1) { if (threadIdx.x < t) red[threadIdx.x] += red[threadIdx.x + t]; __syncthreads(); }
  const float mean = red[0] / D; __syncthreads();
  float vs = 0.f;
  for (int i = threadIdx.x; i < D; i += 128) { float v = xr[i] + (x2r ? x2r[i] : 0.f) - mean; vs += v * v; }
  red[threadIdx.x] = vs; __syncthreads();
  for (int t = 64; t > 0; t >>= 1) { if (threadIdx.x < t) red[threadIdx.x] += red[threadIdx.x + t]; __syncthreads(); }
  const float inv = rsqrtf(red[0] / D + 1e-5f);
  for (int i = threadIdx.x; i < D; i += 128) {
    float v = xr[i] + (x2r ? x2r[i] : 0.f);
    y[row * D + i] = (v - mean) * inv * g[i] + b[i];
  }
}

// mean over rows of length len: out[row] = mean(x[row, :])
__launch_bounds__(128)
__global__ void k_avg(const float* x, float* out, int len) {
  __shared__ float red[128];
  const long long row = blockIdx.x;
  float s = 0.f;
  for (int i = threadIdx.x; i < len; i += 128) s += x[row * len + i];
  red[threadIdx.x] = s; __syncthreads();
  for (int t = 64; t > 0; t >>= 1) { if (threadIdx.x < t) red[threadIdx.x] += red[threadIdx.x + t]; __syncthreads(); }
  if (threadIdx.x == 0) out[row] = red[0] / len;
}

// concat 4x[B,128,1000] -> xcat [B,512,1000]
__global__ void k_concat4(const float* a, const float* b_, const float* c_,
                          const float* d_, float* out) {
  long long idx = (long long)blockIdx.x * blockDim.x + threadIdx.x;
  if (idx >= 8LL * 512 * 1000) return;
  int n = (int)(idx % 1000); int ch = (int)((idx / 1000) % 512); int bb = (int)(idx / 512000);
  const float* s = ch < 128 ? a : ch < 256 ? b_ : ch < 384 ? c_ : d_;
  out[idx] = s[(long long)bb * 128000 + (long long)(ch & 127) * 1000 + n];
}

// im2col for 3x3x3 conv, pad 1, on 10^3 grid; in(b,ci,spatial) via strides
// out[((b*1000+n)*Cin + ci)*27 + t]
__global__ void k_im2col(const float* in, float* out, long long inB, long long inC,
                         long long inS, int Cin, long long total) {
  const long long stride = (long long)gridDim.x * blockDim.x;
  for (long long idx = (long long)blockIdx.x * blockDim.x + threadIdx.x; idx < total; idx += stride) {
    int t = (int)(idx % 27); long long q = idx / 27;
    int ci = (int)(q % Cin); q /= Cin;
    int n = (int)(q % 1000); int b = (int)(q / 1000);
    int x = n % 10, y = (n / 10) % 10, z = n / 100;
    int xx = x + t % 3 - 1, yy = y + (t / 3) % 3 - 1, zz = z + t / 9 - 1;
    float v = 0.f;
    if ((unsigned)xx < 10u && (unsigned)yy < 10u && (unsigned)zz < 10u)
      v = in[(long long)b * inB + (long long)ci * inC + (long long)(zz * 100 + yy * 10 + xx) * inS];
    out[idx] = v;
  }
}

// x_lin[(b*1000+n)*512+c] = xcat[b,c,n] + avg[b,c] + pos[c,n]
__global__ void k_xlin(const float* xcat, const float* avg, const float* pos, float* out) {
  long long idx = (long long)blockIdx.x * blockDim.x + threadIdx.x;
  if (idx >= 8000LL * 512) return;
  int c = (int)(idx % 512); long long m = idx / 512;
  int n = (int)(m % 1000); int b = (int)(m / 1000);
  out[idx] = xcat[(long long)b * 512000 + (long long)c * 1000 + n] + avg[b * 512 + c]
           + pos[(long long)c * 1000 + n];
}

// patch embed: [B,Cc,1000] -> [B*1000, Cc]
__global__ void k_transpose(const float* in, float* out, long long total, int Cc) {
  long long idx = (long long)blockIdx.x * blockDim.x + threadIdx.x;
  if (idx >= total) return;
  int c = (int)(idx % Cc); long long m = idx / Cc;
  int n = (int)(m % 1000); int b = (int)(m / 1000);
  out[idx] = in[(long long)b * Cc * 1000 + (long long)c * 1000 + n];
}

__global__ void k_zero(float* p, long long total) {
  long long idx = (long long)blockIdx.x * blockDim.x + threadIdx.x;
  if (idx < total) p[idx] = 0.f;
}

// split packed qkv [m, 3*(h*16+d)+s] into F_Q / F_K columns, accumulate F_V
__launch_bounds__(128)
__global__ void k_extract(const float* qkv, float* FQ, float* FK, float* FV, int mod) {
  const long long m = blockIdx.x; const int c = threadIdx.x;     // grid 8000 x 128
  const float q = qkv[m * 384 + 3 * c];
  const float k = qkv[m * 384 + 3 * c + 1];
  const float v = qkv[m * 384 + 3 * c + 2];
  FQ[m * 512 + mod * 128 + c] = q;
  FK[m * 512 + mod * 128 + c] = k;
  FV[m * 128 + c] += v;
}

// MFI 0.7/0.3 blends (Q,K from packed qk [m, 2c+s]; V from v buffer)
__launch_bounds__(256)
__global__ void k_blend(const float* qk, const float* vb, float* Q, float* K, float* V) {
  const long long m = blockIdx.x; const int c = threadIdx.x;     // grid 8000 x 256
  const long long i = m * 256 + c;
  Q[i] = 0.7f * qk[m * 512 + 2 * c]     + 0.3f * Q[i];
  K[i] = 0.7f * qk[m * 512 + 2 * c + 1] + 0.3f * K[i];
  V[i] = 0.3f * V[i] + 0.7f * vb[i];
}

// ---------------------------------------------------------------------------
// Host orchestration
// ---------------------------------------------------------------------------
extern "C" void kernel_launch(void* const* d_in, const int* in_sizes, int n_in,
                              void* d_out, int out_size, void* d_ws, size_t ws_size,
                              hipStream_t stream) {
  (void)in_sizes; (void)n_in; (void)out_size; (void)ws_size;
  auto F = [&](int i) { return (const float*)d_in[i]; };

  // jax pytree flatten order (dict keys sorted alphabetically at every level):
  // 0: flair input; 1..183: params{flair,lk,lq,lv,mfc,mfi,t1,t1ce,t2}; 184..186: t1,t1ce,t2
  struct Mode { const float *ln_b[4], *ln_g[4], *out_b[4], *out_w[4], *qkv_b[4], *qkv_w[4]; };
  auto rdMode = [&](int base) {
    Mode m;
    for (int b = 0; b < 4; ++b) {           // block keys sorted: ln_b,ln_g,out_b,out_w,qkv_b,qkv_w
      m.ln_b[b]  = F(base + 6 * b + 0); m.ln_g[b]  = F(base + 6 * b + 1);
      m.out_b[b] = F(base + 6 * b + 2); m.out_w[b] = F(base + 6 * b + 3);
      m.qkv_b[b] = F(base + 6 * b + 4); m.qkv_w[b] = F(base + 6 * b + 5);
    }
    return m;
  };
  struct Fus { const float *b0, *b1, *b2, *ln_b, *ln_g, *w0, *w1, *w2; };
  auto rdFus = [&](int base) {              // keys sorted: b0,b1,b2,ln_b,ln_g,w0,w1,w2
    Fus f; f.b0 = F(base); f.b1 = F(base + 1); f.b2 = F(base + 2); f.ln_b = F(base + 3);
    f.ln_g = F(base + 4); f.w0 = F(base + 5); f.w1 = F(base + 6); f.w2 = F(base + 7); return f;
  };
  const float* img_flair = F(0);
  Mode pm_flair = rdMode(1);
  Fus  lk = rdFus(25), lq = rdFus(33), lv = rdFus(41);
  // mfc keys sorted: lin_b,lin_w,ln1_b,ln1_g,ln2_b,ln2_g,pos,res{b1,b2,bs,w1,w2,ws}
  const float* mfc_lin_b = F(49); const float* mfc_lin_w = F(50);
  const float* mfc_ln1_b = F(51); const float* mfc_ln1_g = F(52);
  const float* mfc_ln2_b = F(53); const float* mfc_ln2_g = F(54);
  const float* mfc_pos   = F(55);
  const float* res_b1 = F(56); const float* res_b2 = F(57); const float* res_bs = F(58);
  const float* res_w1 = F(59); const float* res_w2 = F(60); const float* res_ws = F(61);
  // mfi keys sorted: blocks (4 x sorted{ln_b,ln_g,m1_b,m1_w,m2_b,m2_w,out_b,out_w,qk_b,qk_w,v_b,v_w}), ln_b, ln_g
  struct MfiB { const float *ln_b, *ln_g, *m1_b, *m1_w, *m2_b, *m2_w, *out_b, *out_w, *qk_b, *qk_w, *v_b, *v_w; };
  MfiB mfi[4];
  for (int b = 0; b < 4; ++b) {
    int s = 62 + 12 * b;
    mfi[b] = { F(s), F(s + 1), F(s + 2), F(s + 3), F(s + 4), F(s + 5),
               F(s + 6), F(s + 7), F(s + 8), F(s + 9), F(s + 10), F(s + 11) };
  }
  const float* mfi_ln_b = F(110); const float* mfi_ln_g = F(111);
  Mode pm_t1 = rdMode(112), pm_t1ce = rdMode(136), pm_t2 = rdMode(160);
  const float* img_t1 = F(184); const float* img_t1ce = F(185); const float* img_t2 = F(186);

  // ---- workspace bump allocator ----
  char* wsp = (char*)d_ws; size_t off = 0;
  auto alloc = [&](long long nfl) {
    float* p = (float*)(wsp + off);
    off += (size_t)(((nfl * 4 + 255) / 256) * 256);
    return p;
  };
  const int Bn = 8, N = 1000, C = 128, MFC = 512, CMP = 256, M = 8000;
  float* big   = alloc((long long)M * 13824);     // im2col scratch
  float* xcat  = alloc((long long)Bn * MFC * N);
  float* avg   = alloc((long long)Bn * MFC);
  float* h1    = alloc((long long)M * CMP);
  float* sbuf  = alloc((long long)M * CMP);
  float* xconv = alloc((long long)M * CMP);
  float* xlin  = alloc((long long)M * MFC);
  float* lnbuf = alloc((long long)M * MFC);
  float* proj  = alloc((long long)M * CMP);
  float* Fx    = alloc((long long)M * CMP);
  float* xm    = alloc((long long)M * C);
  float* qkvb  = alloc((long long)M * 3 * C);
  float* om    = alloc((long long)M * C);
  float* FQ    = alloc((long long)M * MFC);
  float* FK    = alloc((long long)M * MFC);
  float* FV    = alloc((long long)M * C);
  float* tmp0  = alloc((long long)M * MFC);
  float* FQf   = alloc((long long)M * CMP);
  float* FKf   = alloc((long long)M * CMP);
  float* FVf   = alloc((long long)M * CMP);
  float* xn2   = alloc((long long)M * CMP);
  float* qkb2  = alloc((long long)M * 2 * CMP);
  float* vb2   = alloc((long long)M * CMP);
  float* omfi  = alloc((long long)M * CMP);
  float* mlph  = alloc((long long)M * 1024);

  auto gemm = [&](const float* A, long long aB2, long long aB1, long long aR, long long aC,
                  const float* Bm, long long bB2, long long bB1, long long bR, long long bC,
                  float* Cc, long long cB2, long long cB1, long long cR,
                  const float* bias, const float* R, long long rB2, long long rB1, long long rR,
                  int Mm, int Nn, int K, int batches, int bdiv, float alpha, int act) {
    GemmP p{A, Bm, Cc, bias, R, aB2, aB1, aR, aC, bB2, bB1, bR, bC,
            cB2, cB1, cR, rB2, rB1, rR, Mm, Nn, K, bdiv, alpha, act};
    dim3 g((Mm + 63) / 64, (Nn + 31) / 32, batches);
    k_gemm<<<g, dim3(128), 0, stream>>>(p);
  };
  auto gemm_lin = [&](const float* A, const float* W, float* Cc, const float* bias,
                      const float* R, int Mm, int Nn, int K, int act) {
    gemm(A, 0, 0, K, 1, W, 0, 0, Nn, 1, Cc, 0, 0, Nn, bias, R, 0, 0, Nn,
         Mm, Nn, K, 1, 1, 1.f, act);
  };
  auto layernorm = [&](const float* x, const float* x2, const float* g_, const float* b_,
                       float* y, int rows, int D) {
    k_layernorm<<<dim3(rows), dim3(128), 0, stream>>>(x, x2, g_, b_, y, D);
  };
  auto flash = [&](const float* Q, long long qB2, long long qB1, long long qR, long long qC,
                   const float* Kp, long long kB2, long long kB1, long long kR, long long kC,
                   const float* Vp, long long vB2, long long vB1, long long vR, long long vC,
                   float* Op, long long oB2, long long oB1, long long oR,
                   int Nq, int Nk, int hs, int batches, int bdiv, float scale) {
    FlashP p{Q, Kp, Vp, Op, qB2, qB1, qR, qC, kB2, kB1, kR, kC,
             vB2, vB1, vR, vC, oB2, oB1, oR, Nq, Nk, hs, bdiv, scale};
    dim3 g((Nq + 63) / 64, batches);
    k_flash<<<g, dim3(128), 0, stream>>>(p);
  };

  // ================= Phase 1: MFC (conv resblock + linear path) =================
  k_concat4<<<dim3(16384), dim3(256), 0, stream>>>(img_t1, img_t1ce, img_t2, img_flair, xcat);
  k_avg<<<dim3(Bn * MFC), dim3(128), 0, stream>>>(xcat, avg, N);
  k_im2col<<<dim3(32768), dim3(256), 0, stream>>>(xcat, big, (long long)MFC * N, (long long)N, 1LL,
                                                  MFC, (long long)M * 13824);
  // conv1: [M,13824]@w1^T (+b1, relu) ; w1 (O=256, I*27=13824) -> B(k,n)=w1[n*13824+k]
  gemm(big, 0, 0, 13824, 1, res_w1, 0, 0, 1, 13824, h1, 0, 0, 256,
       res_b1, nullptr, 0, 0, 0, M, 256, 13824, 1, 1, 1.f, 1);
  k_im2col<<<dim3(32768), dim3(256), 0, stream>>>(h1, big, (long long)N * CMP, 1LL, (long long)CMP,
                                                  CMP, (long long)M * 6912);
  // 1x1x1 shortcut straight off NCDHW xcat: A(n,c)=xcat[b,c,n]
  gemm(xcat, 0, (long long)MFC * N, 1, N, res_ws, 0, 0, 1, 512, sbuf, 0, (long long)N * CMP, CMP,
       res_bs, nullptr, 0, 0, 0, N, CMP, MFC, Bn, Bn, 1.f, 0);
  // conv2 + shortcut residual + relu -> xconv (token-major == xce)
  gemm(big, 0, 0, 6912, 1, res_w2, 0, 0, 1, 6912, xconv, 0, 0, 256,
       res_b2, sbuf, 0, 0, 256, M, 256, 6912, 1, 1, 1.f, 1);
  // x_lin = patch(xcat) + avg + pos ; LN ; Linear 512->256 ; Fx = LN(xce + x_lin)
  k_xlin<<<dim3(16384), dim3(256), 0, stream>>>(xcat, avg, mfc_pos, xlin);
  layernorm(xlin, nullptr, mfc_ln1_g, mfc_ln1_b, lnbuf, M, MFC);
  gemm_lin(lnbuf, mfc_lin_w, proj, mfc_lin_b, nullptr, M, CMP, MFC, 0);
  layernorm(xconv, proj, mfc_ln2_g, mfc_ln2_b, Fx, M, CMP);

  // ================= Phase 2: mode transformers (4 modalities x 4 blocks) ======
  k_zero<<<dim3(4096), dim3(256), 0, stream>>>(FV, (long long)M * C);
  auto run_mode = [&](const float* img, const Mode& pm, int mod) {
    k_transpose<<<dim3(4096), dim3(256), 0, stream>>>(img, xm, (long long)M * C, C);
    for (int b = 0; b < 4; ++b) {
      layernorm(xm, nullptr, pm.ln_g[b], pm.ln_b[b], lnbuf, M, C);
      gemm_lin(lnbuf, pm.qkv_w[b], qkvb, pm.qkv_b[b], nullptr, M, 3 * C, C, 0);
      // fused attention straight off packed qkv: q/k/v element stride 3, head stride 48
      flash(qkvb,     384000, 48, 384, 3,
            qkvb + 1, 384000, 48, 384, 3,
            qkvb + 2, 384000, 48, 384, 3,
            om, 128000, 16, 128,
            N, N, 16, 64, 8, 0.25f);
      gemm_lin(om, pm.out_w[b], xm, pm.out_b[b], xm, M, C, C, 0);   // x += proj(o)
    }
    k_extract<<<dim3(8000), dim3(128), 0, stream>>>(qkvb, FQ, FK, FV, mod);
  };
  run_mode(img_t1,   pm_t1,   0);
  run_mode(img_t1ce, pm_t1ce, 1);
  run_mode(img_t2,   pm_t2,   2);
  run_mode(img_flair, pm_flair, 3);

  // ================= Phase 3: fusion MLPs ======================================
  auto run_fusion = [&](const float* X, const Fus& f, int din, float* out) {
    layernorm(X, nullptr, f.ln_g, f.ln_b, lnbuf, M, din);
    gemm_lin(lnbuf, f.w0, proj, f.b0, nullptr, M, 256, din, 0);
    gemm_lin(proj, f.w1, tmp0, f.b1, nullptr, M, 512, 256, 2);      // GELU
    gemm_lin(tmp0, f.w2, out, f.b2, nullptr, M, 256, 512, 0);
  };
  run_fusion(FQ, lq, 512, FQf);
  run_fusion(FK, lk, 512, FKf);
  run_fusion(FV, lv, 128, FVf);

  // ================= Phase 4: MFI (4 blocks, hs=32) ============================
  float* X = Fx;
  for (int b = 0; b < 4; ++b) {
    const MfiB& pb = mfi[b];
    layernorm(X, nullptr, pb.ln_g, pb.ln_b, xn2, M, CMP);
    gemm_lin(xn2, pb.qk_w, qkb2, pb.qk_b, nullptr, M, 512, CMP, 0);
    gemm_lin(xn2, pb.v_w,  vb2,  pb.v_b,  nullptr, M, CMP, CMP, 0);
    k_blend<<<dim3(8000), dim3(256), 0, stream>>>(qkb2, vb2, FQf, FKf, FVf);
    // fused attention on blended Q/K/V (head-merged layout, head stride 32)
    flash(FQf, 256000, 32, 256, 1,
          FKf, 256000, 32, 256, 1,
          FVf, 256000, 32, 256, 1,
          omfi, 256000, 32, 256,
          N, N, 32, 64, 8, 0.17677669529663689f);
    gemm_lin(omfi, pb.out_w, X, pb.out_b, X, M, CMP, CMP, 0);       // x += proj(o)
    layernorm(X, nullptr, pb.ln_g, pb.ln_b, xn2, M, CMP);           // same LN params reused (as ref)
    gemm_lin(xn2, pb.m1_w, mlph, pb.m1_b, nullptr, M, 1024, CMP, 2);
    gemm_lin(mlph, pb.m2_w, X, pb.m2_b, X, M, CMP, 1024, 0);        // x += mlp
  }
  layernorm(X, nullptr, mfi_ln_g, mfi_ln_b, (float*)d_out, M, CMP);
}
